// GRU_64424509440218
// MI455X (gfx1250) — matrix operand order repaired and checked by reading, not compiled, not run on hardware
//
#include <hip/hip_runtime.h>

typedef __attribute__((ext_vector_type(16))) _Float16 v16h;
typedef __attribute__((ext_vector_type(8)))  float    v8f;
typedef __attribute__((ext_vector_type(2)))  float    v2f;
typedef __attribute__((ext_vector_type(4)))  unsigned int u32x4;
typedef __attribute__((ext_vector_type(8)))  int      i32x8;
typedef __attribute__((ext_vector_type(4)))  int      i32x4;

#define B_  64
#define T_  2048
#define I_  128
#define H_  256
#define G_  768   // 3*H

// ---------------------------------------------------------------------------
// Kernel 0: pack W_hh (768x256 f32, row-major) into WMMA-B-operand-ready f16
// tiles for V_WMMA_F32_16X16X32_F16.  B is 32x16 (KxN), 8 VGPRs/lane:
//   lanes 0-15 hold K=0..15 (2 halves per VGPR), lanes 16-31 hold K=16..31.
// Tile linear index = (gate*16 + wave)*8 + k0tile; 512 halves (1KB) per tile.
// Gates r,z occupy the first contiguous 256KB -> TDM-staged into LDS once.
// ---------------------------------------------------------------------------
__global__ __launch_bounds__(256)
void pack_whh(const float* __restrict__ W_hh, _Float16* __restrict__ wp)
{
    const int gid  = blockIdx.x * 256 + threadIdx.x;   // 384 tiles * 32 lanes
    const int tile = gid >> 5;
    const int lane = gid & 31;
    const int gate = tile >> 7;        // 0..2
    const int rem  = tile & 127;
    const int wv   = rem >> 3;         // 0..15
    const int k0   = rem & 7;          // 0..7
    const int g    = gate * 256 + wv * 16 + (lane & 15);
    const int kb   = k0 * 32 + 16 * (lane >> 4);
    _Float16* dst  = wp + (size_t)tile * 512 + lane * 16;
#pragma unroll
    for (int s = 0; s < 16; ++s)
        dst[s] = (_Float16)W_hh[g * H_ + kb + s];
}

// ---------------------------------------------------------------------------
// Kernel 1: gi[r][g] = x[r][:] . W_ih[g][:] + b_ih[g],  r = b*T+t  (131072 rows)
// Full-f32 GEMM via V_WMMA_F32_16X16X4_F32.  One block = one 16-row M tile,
// 16 waves cover all 48 N-tiles (3 gate tiles per wave, shared A operand).
// x tile staged in LDS (row stride padded to 132 words -> conflict-free v2f).
// ---------------------------------------------------------------------------
__global__ __launch_bounds__(512)
void gi_gemm(const float* __restrict__ x, const float* __restrict__ W_ih,
             const float* __restrict__ b_ih, float* __restrict__ gi)
{
    __shared__ float xs[16][132];
    const int tid  = threadIdx.x;
    const int lane = tid & 31;
    const int wv   = tid >> 5;          // 0..15
    const int ln   = lane & 15;
    const int hi   = lane >> 4;         // 0 or 1
    const int R0   = blockIdx.x * 16;

    {   // stage x tile: wave w loads row w, 4 floats per lane
        const int c = lane * 4;
        const float4 v = *reinterpret_cast<const float4*>(x + (size_t)(R0 + wv) * I_ + c);
        xs[wv][c + 0] = v.x; xs[wv][c + 1] = v.y;
        xs[wv][c + 2] = v.z; xs[wv][c + 3] = v.w;
    }
    __syncthreads();

    v8f acc0 = {}, acc1 = {}, acc2 = {};
    const float* bp0 = W_ih + (size_t)(0 * 256 + wv * 16 + ln) * I_ + 2 * hi;
    const float* bp1 = W_ih + (size_t)(1 * 256 + wv * 16 + ln) * I_ + 2 * hi;
    const float* bp2 = W_ih + (size_t)(2 * 256 + wv * 16 + ln) * I_ + 2 * hi;

#pragma unroll
    for (int kk = 0; kk < 32; ++kk) {   // K = 128 in steps of 4
        // A 16x4 f32: lanes 0-15 -> K0,K1 ; lanes 16-31 -> K2,K3
        v2f a  = *reinterpret_cast<const v2f*>(&xs[ln][4 * kk + 2 * hi]);
        v2f b0 = *reinterpret_cast<const v2f*>(bp0 + 4 * kk);
        v2f b1 = *reinterpret_cast<const v2f*>(bp1 + 4 * kk);
        v2f b2 = *reinterpret_cast<const v2f*>(bp2 + 4 * kk);
        acc0 = __builtin_amdgcn_wmma_f32_16x16x4_f32(false, a, false, b0, (short)0, acc0, false, false);
        acc1 = __builtin_amdgcn_wmma_f32_16x16x4_f32(false, a, false, b1, (short)0, acc1, false, false);
        acc2 = __builtin_amdgcn_wmma_f32_16x16x4_f32(false, a, false, b2, (short)0, acc2, false, false);
    }

    // C/D layout: VGPR i, lanes 0-15 -> M=i ; lanes 16-31 -> M=i+8 ; N=ln
#pragma unroll
    for (int g = 0; g < 3; ++g) {
        const int   col  = g * 256 + wv * 16 + ln;
        const float bias = b_ih[col];
        const v8f   a    = (g == 0) ? acc0 : (g == 1) ? acc1 : acc2;
#pragma unroll
        for (int i = 0; i < 8; ++i)
            gi[(size_t)(R0 + i + 8 * hi) * G_ + col] = a[i] + bias;
    }
}

// ---------------------------------------------------------------------------
// Kernel 2: sequential GRU scan. grid = 4 independent blocks (16 batches each),
// 512 threads = 16 waves.  Wave w owns gate columns {j, j+256, j+512}.
// r,z gate weights (256KB f16) are TDM-staged into LDS once and stay resident
// for all 2048 steps (CDNA5: 320KB LDS/WG); n-gate streams from L2 (128KB/step)
// so per-step traffic is split across the DS and VMEM pipes.
// Per step: 8 K-tiles x 3 gates = 24 v_wmma_f32_16x16x32_f16; h_old lives in
// registers; h_new scattered into double-buffered f16 A-layout LDS; 1 barrier.
// ---------------------------------------------------------------------------
__global__ __launch_bounds__(512)
void gru_scan(const float* __restrict__ gi, const _Float16* __restrict__ wp,
              const float* __restrict__ b_hh, float* __restrict__ hid)
{
    __shared__ alignas(32) _Float16 ldsW[2 * 16 * 8 * 512];   // 256 KB: gates r,z
    __shared__ alignas(32) _Float16 hA[2][8][32][16];         // 16 KB, double buffered
    const int tid  = threadIdx.x;
    const int lane = tid & 31;
    const int wv   = tid >> 5;
    const int ln   = lane & 15;
    const int hi   = lane >> 4;
    const int j    = wv * 16 + ln;          // this thread's hidden column
    const int bg   = blockIdx.x;            // batch group 0..3

    {   // zero both A buffers (h0 = 0)
        uint32_t* z = (uint32_t*)&hA[0][0][0][0];
        for (int k = tid; k < (int)(sizeof(hA) / 4); k += 512) z[k] = 0u;
    }

    // --- TDM: one 1-D descriptor copies the first 256KB of packed weights
    //     (gates r,z) into LDS.  32768 elements of 8 bytes.
    if (tid < 32) {
        const unsigned long long ga = (unsigned long long)(uintptr_t)wp;
        const unsigned int       la = (unsigned int)(uintptr_t)&ldsW[0];
        u32x4 g0 = { 1u,                                   // count=1, user mode
                     la,                                   // lds_addr
                     (unsigned int)(ga & 0xffffffffu),     // global_addr[31:0]
                     (unsigned int)((ga >> 32) & 0x01ffffffu) | 0x80000000u }; // [56:32] | type=2
        i32x8 g1 = { (int)(3u << 16),      // data_size = 3 (8 bytes)
                     (int)0x80000000,      // tensor_dim0[15:0]=32768 in [31:16]
                     (int)0x00010000,      // tensor_dim0[31:16]=0; tensor_dim1=1
                     (int)0x80000000,      // tile_dim0=32768 in [31:16]
                     0,                    // tile_dim1=0, tile_dim2=0 (unused)
                     0x00008000,           // tensor_dim0_stride[31:0]=32768
                     0, 0 };               // stride hi / dim1_stride = 0
        i32x4 g2 = { 0, 0, 0, 0 };
        i32x4 g3 = { 0, 0, 0, 0 };
        i32x8 g4 = { 0, 0, 0, 0, 0, 0, 0, 0 };   // extra group (6-arg toolchain)
        __builtin_amdgcn_tensor_load_to_lds(g0, g1, g2, g3, g4, 0);
        __builtin_amdgcn_s_wait_tensorcnt(0);
    }

    float hreg[8];
#pragma unroll
    for (int i = 0; i < 8; ++i) hreg[i] = 0.f;

    const float br = b_hh[j], bz = b_hh[256 + j], bn = b_hh[512 + j];

    size_t giBase[8], hBase[8];
#pragma unroll
    for (int i = 0; i < 8; ++i) {
        const int b = bg * 16 + i + 8 * hi;         // C-layout row -> batch
        giBase[i] = (size_t)b * T_ * G_ + j;
        hBase[i]  = (size_t)b * T_ * H_ + j;
    }

    // LDS-resident B pointers (r,z) and global B pointer (n)
    const _Float16* lWr = ldsW + ((size_t)(0 * 16 + wv) * 8) * 512 + lane * 16;
    const _Float16* lWz = ldsW + ((size_t)(1 * 16 + wv) * 8) * 512 + lane * 16;
    const _Float16* wpn = wp   + ((size_t)(2 * 16 + wv) * 8) * 512 + lane * 16;

    // h -> f16 A-operand scatter mapping (A 16x32, contiguous k order matching packed B):
    // lanes 0-15: slots 0-15 => K0-15
    // lanes16-31: slots 0-15 => K16-31
    const int k0t = j >> 5;
    const int kk  = j & 31;
    int wl, wsl;
    if      (kk < 16) { wl = 0;  wsl = kk;      }
    else              { wl = 16; wsl = kk - 16; }

    __syncthreads();

    int p = 0;
    for (int t = 0; t < T_; ++t) {
        v8f cr = {}, cz = {}, cn = {};
#pragma unroll
        for (int k0 = 0; k0 < 8; ++k0) {            // K = 256 in steps of 32
            v16h a  = *reinterpret_cast<const v16h*>(&hA[p][k0][lane][0]);
            v16h wr = *reinterpret_cast<const v16h*>(lWr + k0 * 512);
            v16h wz = *reinterpret_cast<const v16h*>(lWz + k0 * 512);
            v16h wn = *reinterpret_cast<const v16h*>(wpn + k0 * 512);
            cr = __builtin_amdgcn_wmma_f32_16x16x32_f16(false, a, false, wr, (short)0, cr, false, false);
            cz = __builtin_amdgcn_wmma_f32_16x16x32_f16(false, a, false, wz, (short)0, cz, false, false);
            cn = __builtin_amdgcn_wmma_f32_16x16x32_f16(false, a, false, wn, (short)0, cn, false, false);
        }

#pragma unroll
        for (int i = 0; i < 8; ++i) {
            const size_t o  = giBase[i] + (size_t)t * G_;
            const float  ir = gi[o];
            const float  iz = gi[o + 256];
            const float  in = gi[o + 512];
            const float  r  = __builtin_amdgcn_rcpf(1.f + __expf(-(ir + cr[i] + br)));
            const float  z  = __builtin_amdgcn_rcpf(1.f + __expf(-(iz + cz[i] + bz)));
            const float  gn = in + r * (cn[i] + bn);
            const float  e  = __expf(-2.f * gn);
            const float  n  = (1.f - e) * __builtin_amdgcn_rcpf(1.f + e);
            const float  h  = (1.f - z) * n + z * hreg[i];
            hreg[i] = h;
            hid[hBase[i] + (size_t)t * H_] = h;                 // [B,T,H] output
            hA[1 - p][k0t][wl + i + 8 * hi][wsl] = (_Float16)h; // next step's A
        }
        p ^= 1;
        __syncthreads();
    }
}

// ---------------------------------------------------------------------------
// Kernel 3: outputs[r] = hs[r][:] . W_o + b_o   (one wave per row)
// ---------------------------------------------------------------------------
__global__ __launch_bounds__(256)
void out_proj(const float* __restrict__ hid, const float* __restrict__ W_o,
              const float* __restrict__ b_o, float* __restrict__ out)
{
    const int lane = threadIdx.x & 31;
    const int row  = blockIdx.x * 8 + (threadIdx.x >> 5);
    float p = 0.f;
#pragma unroll
    for (int c = 0; c < 8; ++c)
        p += hid[(size_t)row * H_ + lane + 32 * c] * W_o[lane + 32 * c];
#pragma unroll
    for (int off = 16; off >= 1; off >>= 1)
        p += __shfl_xor(p, off, 32);
    if (lane == 0) out[row] = p + b_o[0];
}

// ---------------------------------------------------------------------------
extern "C" void kernel_launch(void* const* d_in, const int* in_sizes, int n_in,
                              void* d_out, int out_size, void* d_ws, size_t ws_size,
                              hipStream_t stream)
{
    const float* x    = (const float*)d_in[0];
    const float* W_ih = (const float*)d_in[1];
    const float* W_hh = (const float*)d_in[2];
    const float* b_ih = (const float*)d_in[3];
    const float* b_hh = (const float*)d_in[4];
    const float* W_o  = (const float*)d_in[5];
    const float* b_o  = (const float*)d_in[6];

    float* out = (float*)d_out;                 // [B,T,1]
    float* hid = out + (size_t)B_ * T_;         // [B,T,H]

    float*    gi = (float*)d_ws;                                        // 384 MB
    _Float16* wp = (_Float16*)((char*)d_ws + (size_t)B_ * T_ * G_ * 4); // 384 KB

    pack_whh<<<48, 256, 0, stream>>>(W_hh, wp);
    gi_gemm<<<(B_ * T_) / 16, 512, 0, stream>>>(x, W_ih, b_ih, gi);
    gru_scan<<<4, 512, 0, stream>>>(gi, wp, b_hh, hid);
    out_proj<<<(B_ * T_) / 8, 256, 0, stream>>>(hid, W_o, b_o, out);
}
